// Hybrid_SelfAttention_MRHA3_13615046328327
// MI455X (gfx1250) — compile-verified, matrix operand order checked by reading, not run on hardware
//
#include <hip/hip_runtime.h>
#include <hip/hip_bf16.h>
#include <math.h>

typedef __attribute__((ext_vector_type(16))) _Float16 v16h;
typedef __attribute__((ext_vector_type(8)))  _Float16 v8h;
typedef __attribute__((ext_vector_type(8)))  float    v8f;

#define Bn  8
#define Tn  2048
#define Tdn 1024
#define Cn  512

struct Frag { union { v16h v; v8h h[2]; }; };

__device__ __forceinline__ void load_frag(Frag& f, const _Float16* p) {
    f.h[0] = *(const v8h*)(p);
    f.h[1] = *(const v8h*)(p + 16);
}

// ---------------------------------------------------------------------------
// f32 -> f16 conversion
// ---------------------------------------------------------------------------
__global__ void f2h_kernel(const float* __restrict__ src, _Float16* __restrict__ dst, size_t n) {
    size_t i = (size_t)blockIdx.x * blockDim.x + threadIdx.x;
    if (i < n) dst[i] = (_Float16)src[i];
}

// ---------------------------------------------------------------------------
// Downsample Conv1d(k=3, stride=2, pad=1).
// x: [B,T,C] f32 ; dw: [Cout,Cin,3] ; out: [B,Td,C] f16
// Input rows staged into LDS with GLOBAL_LOAD_ASYNC_TO_LDS_B128 (ASYNCcnt path).
// grid: (C/256, Td, B), block 256
// ---------------------------------------------------------------------------
__global__ void conv_down_kernel(const float* __restrict__ x, const float* __restrict__ dw,
                                 const float* __restrict__ db, _Float16* __restrict__ xdh) {
    __shared__ float r[3][Cn];
    int b = blockIdx.z, td = blockIdx.y;
    int tid = threadIdx.x;

    // 3 rows * 512 f32 = 384 x 16B chunks; async-copy global -> LDS
    for (int it = 0; it < 2; ++it) {
        int i = tid + it * 256;
        if (i < 384) {
            int kk = i >> 7;           // row 0..2
            int c4 = i & 127;          // 16B chunk within row
            int t  = 2 * td - 1 + kk;
            unsigned ldsoff = (unsigned)(size_t)&r[kk][c4 * 4];
            if (t >= 0 && t < Tn) {
                unsigned long long ga =
                    (unsigned long long)(x + ((size_t)b * Tn + t) * Cn + c4 * 4);
                asm volatile("global_load_async_to_lds_b128 %0, %1, off"
                             :: "v"(ldsoff), "v"(ga) : "memory");
            } else {
                float4 z = {0.f, 0.f, 0.f, 0.f};
                *(float4*)&r[kk][c4 * 4] = z;
            }
        }
    }
    asm volatile("s_wait_asynccnt 0x0" ::: "memory");
    __syncthreads();

    int co = blockIdx.x * 256 + tid;
    float acc = db[co];
    const float* w = dw + (size_t)co * Cn * 3;
    for (int ci = 0; ci < Cn; ++ci) {
        acc += w[ci * 3 + 0] * r[0][ci];
        acc += w[ci * 3 + 1] * r[1][ci];
        acc += w[ci * 3 + 2] * r[2][ci];
    }
    xdh[((size_t)b * Tdn + td) * Cn + co] = (_Float16)acc;
}

// ---------------------------------------------------------------------------
// Generic batched WMMA GEMM:  Y[m,n] = scale * sum_k A[m,k]*Bm[n,k]  (+ bias[n])
// A: [M,K] f16 row-major (row stride sA), Bm: [N,K] f16 row-major (row stride sB).
// Per-wave tile: 16(M) x 64(N); K stepped by 32, software-pipelined double buffer.
// Requires: K multiple of 64, M mult 16, N mult 64.
// If tempPtr != nullptr: scale = 1 / max(*tempPtr, 0.01).
// ---------------------------------------------------------------------------
__global__ __launch_bounds__(256)
void gemm_wmma_f16(const _Float16* __restrict__ A, int sA, long aBat,
                   const _Float16* __restrict__ Bm, int sB, long bBat,
                   float* __restrict__ Yf, _Float16* __restrict__ Yh, int sY, long yBat,
                   const float* __restrict__ bias,
                   float constScale, const float* __restrict__ tempPtr,
                   int M, int N, int K) {
    int b    = blockIdx.z;
    int lane = threadIdx.x & 31;
    int wid  = threadIdx.x >> 5;
    int nG   = N >> 6;
    int tile = blockIdx.x * 8 + wid;
    if (tile >= (M >> 4) * nG) return;
    int mt = tile / nG, ng = tile % nG;
    int m0 = mt << 4, n0 = ng << 6;

    float scale = constScale;
    if (tempPtr) scale = 1.0f / fmaxf(*tempPtr, 0.01f);

    const _Float16* Ab = A  + (size_t)b * aBat;
    const _Float16* Bb = Bm + (size_t)b * bBat;

    // WMMA 16x32 f16 A/B-tile layout: lane L holds row (base + (L&15));
    // K-halves selected by lane group (L>>4).
    int lr = lane & 15;
    int kb = (lane >> 4) << 3;
    const _Float16* pa = Ab + (size_t)(m0 + lr) * sA + kb;
    const _Float16* pb0 = Bb + (size_t)(n0 +  0 + lr) * sB + kb;
    const _Float16* pb1 = Bb + (size_t)(n0 + 16 + lr) * sB + kb;
    const _Float16* pb2 = Bb + (size_t)(n0 + 32 + lr) * sB + kb;
    const _Float16* pb3 = Bb + (size_t)(n0 + 48 + lr) * sB + kb;

    const v8f vzero = {0.f, 0.f, 0.f, 0.f, 0.f, 0.f, 0.f, 0.f};
    v8f acc0 = vzero, acc1 = vzero, acc2 = vzero, acc3 = vzero;

    // Double-buffered fragments (distinct registers -> loads of step s+1 overlap
    // with the 4 WMMAs of step s).
    Frag aA, aB, b0A, b1A, b2A, b3A, b0B, b1B, b2B, b3B;

#define LOAD_SET_A(k0) do { \
        load_frag(aA, pa + (k0));  \
        load_frag(b0A, pb0 + (k0)); load_frag(b1A, pb1 + (k0)); \
        load_frag(b2A, pb2 + (k0)); load_frag(b3A, pb3 + (k0)); \
    } while (0)
#define LOAD_SET_B(k0) do { \
        load_frag(aB, pa + (k0));  \
        load_frag(b0B, pb0 + (k0)); load_frag(b1B, pb1 + (k0)); \
        load_frag(b2B, pb2 + (k0)); load_frag(b3B, pb3 + (k0)); \
    } while (0)
#define COMPUTE_A() do { \
        acc0 = __builtin_amdgcn_wmma_f32_16x16x32_f16(false, aA.v, false, b0A.v, (short)0, acc0, false, false); \
        acc1 = __builtin_amdgcn_wmma_f32_16x16x32_f16(false, aA.v, false, b1A.v, (short)0, acc1, false, false); \
        acc2 = __builtin_amdgcn_wmma_f32_16x16x32_f16(false, aA.v, false, b2A.v, (short)0, acc2, false, false); \
        acc3 = __builtin_amdgcn_wmma_f32_16x16x32_f16(false, aA.v, false, b3A.v, (short)0, acc3, false, false); \
    } while (0)
#define COMPUTE_B() do { \
        acc0 = __builtin_amdgcn_wmma_f32_16x16x32_f16(false, aB.v, false, b0B.v, (short)0, acc0, false, false); \
        acc1 = __builtin_amdgcn_wmma_f32_16x16x32_f16(false, aB.v, false, b1B.v, (short)0, acc1, false, false); \
        acc2 = __builtin_amdgcn_wmma_f32_16x16x32_f16(false, aB.v, false, b2B.v, (short)0, acc2, false, false); \
        acc3 = __builtin_amdgcn_wmma_f32_16x16x32_f16(false, aB.v, false, b3B.v, (short)0, acc3, false, false); \
    } while (0)

    int steps = K >> 5;        // even (>= 16) for all call sites
    LOAD_SET_A(0);
    int s = 0;
    for (; s + 2 < steps; s += 2) {
        LOAD_SET_B((s + 1) << 5);
        __builtin_prefetch(pa + ((s + 3) << 5), 0, 3);   // global_prefetch_b8
        COMPUTE_A();
        LOAD_SET_A((s + 2) << 5);
        COMPUTE_B();
    }
    LOAD_SET_B((s + 1) << 5);
    COMPUTE_A();
    COMPUTE_B();

#undef LOAD_SET_A
#undef LOAD_SET_B
#undef COMPUTE_A
#undef COMPUTE_B

    // C/D layout: lane L -> N = n0+16j+(L&15), VGPR v -> M = m0 + v + 8*(L>>4)
    int lh = lane >> 4;
    v8f accs[4] = {acc0, acc1, acc2, acc3};
#pragma unroll
    for (int j = 0; j < 4; ++j) {
#pragma unroll
        for (int v = 0; v < 8; ++v) {
            int m = m0 + v + (lh << 3);
            int n = n0 + (j << 4) + lr;
            float val = accs[j][v] * scale + (bias ? bias[n] : 0.0f);
            size_t yo = (size_t)b * yBat + (size_t)m * sY + n;
            if (Yf) Yf[yo] = val;
            if (Yh) Yh[yo] = (_Float16)val;
        }
    }
}

// ---------------------------------------------------------------------------
// Row softmax over S f32 logits; writes f16 probabilities in place at row start.
// grid: B*T rows, block 256.  S <= 2048.
// ---------------------------------------------------------------------------
__global__ void softmax_rows_kernel(float* __restrict__ L, int S) {
    __shared__ float row[2048];
    __shared__ float red[256];
    int tid = threadIdx.x;
    float* Lr = L + (size_t)blockIdx.x * S;

    float lmax = -3.4e38f;
    for (int i = tid; i < S; i += 256) { float x = Lr[i]; row[i] = x; lmax = fmaxf(lmax, x); }
    red[tid] = lmax; __syncthreads();
    for (int off = 128; off > 0; off >>= 1) {
        if (tid < off) red[tid] = fmaxf(red[tid], red[tid + off]);
        __syncthreads();
    }
    float m = red[0]; __syncthreads();

    float lsum = 0.0f;
    for (int i = tid; i < S; i += 256) { float e = expf(row[i] - m); row[i] = e; lsum += e; }
    red[tid] = lsum; __syncthreads();
    for (int off = 128; off > 0; off >>= 1) {
        if (tid < off) red[tid] += red[tid + off];
        __syncthreads();
    }
    float inv = 1.0f / red[0]; __syncthreads();

    _Float16* P = (_Float16*)Lr;
    for (int i = tid; i < S; i += 256) P[i] = (_Float16)(row[i] * inv);
}

// ---------------------------------------------------------------------------
// f16 transpose per batch: Vt[b,c,s] = V[b,s,c].  grid (S/32, C/32, B), block (32,8)
// ---------------------------------------------------------------------------
__global__ void transpose_f16_kernel(const _Float16* __restrict__ V, _Float16* __restrict__ Vt, int S) {
    __shared__ _Float16 tile[32][33];
    int b = blockIdx.z;
    int s0 = blockIdx.x * 32, c0 = blockIdx.y * 32;
    const _Float16* Vb = V + (size_t)b * S * Cn;
    _Float16* Vtb = Vt + (size_t)b * Cn * S;
#pragma unroll
    for (int k = 0; k < 4; ++k) {
        int s = s0 + threadIdx.y + 8 * k;
        tile[threadIdx.y + 8 * k][threadIdx.x] = Vb[(size_t)s * Cn + c0 + threadIdx.x];
    }
    __syncthreads();
#pragma unroll
    for (int k = 0; k < 4; ++k) {
        int c = c0 + threadIdx.y + 8 * k;
        Vtb[(size_t)c * S + s0 + threadIdx.x] = tile[threadIdx.x][threadIdx.y + 8 * k];
    }
}

// ---------------------------------------------------------------------------
// L2-normalize rows of an f16 [rows, C] matrix in place (wave per row).
// ---------------------------------------------------------------------------
__global__ void l2norm_rows_kernel(_Float16* __restrict__ Q, int rows) {
    int idx  = blockIdx.x * (blockDim.x >> 5) + (threadIdx.x >> 5);
    int lane = threadIdx.x & 31;
    if (idx >= rows) return;
    _Float16* qr = Q + (size_t)idx * Cn;
    float v[16], s = 0.0f;
#pragma unroll
    for (int i = 0; i < 16; ++i) { v[i] = (float)qr[lane * 16 + i]; s += v[i] * v[i]; }
#pragma unroll
    for (int off = 16; off > 0; off >>= 1) s += __shfl_xor(s, off, 32);
    float inv = 1.0f / (sqrtf(s) + 1e-8f);
#pragma unroll
    for (int i = 0; i < 16; ++i) qr[lane * 16 + i] = (_Float16)(v[i] * inv);
}

// ---------------------------------------------------------------------------
// LayerNorm rows: Y = (X-mu)*rsqrt(var+eps)*g + b  (wave per row, C=512)
// ---------------------------------------------------------------------------
__global__ void ln_rows_kernel(const float* __restrict__ X, float* __restrict__ Y,
                               const float* __restrict__ g, const float* __restrict__ bb, int rows) {
    int idx  = blockIdx.x * (blockDim.x >> 5) + (threadIdx.x >> 5);
    int lane = threadIdx.x & 31;
    if (idx >= rows) return;
    const float* xr = X + (size_t)idx * Cn;
    float v[16], s = 0.0f;
#pragma unroll
    for (int i = 0; i < 16; ++i) { v[i] = xr[lane * 16 + i]; s += v[i]; }
#pragma unroll
    for (int off = 16; off > 0; off >>= 1) s += __shfl_xor(s, off, 32);
    float mu = s * (1.0f / Cn);
    float vs = 0.0f;
#pragma unroll
    for (int i = 0; i < 16; ++i) { float d = v[i] - mu; vs += d * d; }
#pragma unroll
    for (int off = 16; off > 0; off >>= 1) vs += __shfl_xor(vs, off, 32);
    float inv = rsqrtf(vs * (1.0f / Cn) + 1e-5f);
    float* yr = Y + (size_t)idx * Cn;
#pragma unroll
    for (int i = 0; i < 16; ++i) {
        int c = lane * 16 + i;
        yr[c] = (v[i] - mu) * inv * g[c] + bb[c];
    }
}

// ---------------------------------------------------------------------------
// Gating fusion: per row, gate = softmax(concat(o0,o1,o2) @ gw^T + gb) over 3,
// z = g0*o0 + g1*o1 + g2*o2.  grid = B*T rows, block 128.
// ---------------------------------------------------------------------------
__global__ void gating_kernel(const float* __restrict__ o0, const float* __restrict__ o1,
                              const float* __restrict__ o2, const float* __restrict__ gw,
                              const float* __restrict__ gb, float* __restrict__ z) {
    __shared__ float red[3][128];
    __shared__ float gate[3];
    size_t base = (size_t)blockIdx.x * Cn;
    int tid = threadIdx.x;
    float p0 = 0.f, p1 = 0.f, p2 = 0.f;
    for (int c = tid; c < Cn; c += 128) {
        float a = o0[base + c], b = o1[base + c], d = o2[base + c];
        p0 += a * gw[0 * 3 * Cn + c] + b * gw[0 * 3 * Cn + Cn + c] + d * gw[0 * 3 * Cn + 2 * Cn + c];
        p1 += a * gw[1 * 3 * Cn + c] + b * gw[1 * 3 * Cn + Cn + c] + d * gw[1 * 3 * Cn + 2 * Cn + c];
        p2 += a * gw[2 * 3 * Cn + c] + b * gw[2 * 3 * Cn + Cn + c] + d * gw[2 * 3 * Cn + 2 * Cn + c];
    }
    red[0][tid] = p0; red[1][tid] = p1; red[2][tid] = p2;
    __syncthreads();
    for (int off = 64; off > 0; off >>= 1) {
        if (tid < off) {
            red[0][tid] += red[0][tid + off];
            red[1][tid] += red[1][tid + off];
            red[2][tid] += red[2][tid + off];
        }
        __syncthreads();
    }
    if (tid == 0) {
        float l0 = red[0][0] + gb[0], l1 = red[1][0] + gb[1], l2 = red[2][0] + gb[2];
        float m = fmaxf(l0, fmaxf(l1, l2));
        float e0 = expf(l0 - m), e1 = expf(l1 - m), e2 = expf(l2 - m);
        float inv = 1.0f / (e0 + e1 + e2);
        gate[0] = e0 * inv; gate[1] = e1 * inv; gate[2] = e2 * inv;
    }
    __syncthreads();
    float g0 = gate[0], g1 = gate[1], g2 = gate[2];
    for (int c = tid; c < Cn; c += 128)
        z[base + c] = g0 * o0[base + c] + g1 * o1[base + c] + g2 * o2[base + c];
}

// ---------------------------------------------------------------------------
// Host launcher
// ---------------------------------------------------------------------------
extern "C" void kernel_launch(void* const* d_in, const int* in_sizes, int n_in,
                              void* d_out, int out_size, void* d_ws, size_t ws_size,
                              hipStream_t stream) {
    const float* x     = (const float*)d_in[0];
    const float* dw    = (const float*)d_in[1];
    const float* db    = (const float*)d_in[2];
    const float* W[9]  = {(const float*)d_in[3],  (const float*)d_in[5],  (const float*)d_in[7],
                          (const float*)d_in[11], (const float*)d_in[13], (const float*)d_in[15],
                          (const float*)d_in[19], (const float*)d_in[21], (const float*)d_in[23]};
    const float* Bi[9] = {(const float*)d_in[4],  (const float*)d_in[6],  (const float*)d_in[8],
                          (const float*)d_in[12], (const float*)d_in[14], (const float*)d_in[16],
                          (const float*)d_in[20], (const float*)d_in[22], (const float*)d_in[24]};
    const float* ln_g[3] = {(const float*)d_in[9],  (const float*)d_in[17], (const float*)d_in[25]};
    const float* ln_b[3] = {(const float*)d_in[10], (const float*)d_in[18], (const float*)d_in[26]};
    const float* temp = (const float*)d_in[27];
    const float* gw   = (const float*)d_in[28];
    const float* gb   = (const float*)d_in[29];
    float* zout = (float*)d_out;
    (void)in_sizes; (void)n_in; (void)out_size; (void)ws_size;

    char* ws = (char*)d_ws;
    size_t off = 0;
    auto alloc = [&](size_t bytes) { size_t r = off; off += (bytes + 255) & ~(size_t)255; return r; };

    const size_t BTC  = (size_t)Bn * Tn * Cn;
    const size_t BTdC = (size_t)Bn * Tdn * Cn;
    const size_t CC   = (size_t)Cn * Cn;

    _Float16* XH  = (_Float16*)(ws + alloc(BTC * 2));
    _Float16* XDH = (_Float16*)(ws + alloc(BTdC * 2));
    _Float16* WH  = (_Float16*)(ws + alloc(9 * CC * 2));
    _Float16* QCR = (_Float16*)(ws + alloc(BTC * 2));
    _Float16* KCR = (_Float16*)(ws + alloc(BTdC * 2));
    _Float16* VCR = (_Float16*)(ws + alloc(BTdC * 2));
    _Float16* QD  = (_Float16*)(ws + alloc(BTC * 2));
    _Float16* KD  = (_Float16*)(ws + alloc(BTC * 2));
    _Float16* VD  = (_Float16*)(ws + alloc(BTC * 2));
    _Float16* QC  = (_Float16*)(ws + alloc(BTC * 2));
    _Float16* KC  = (_Float16*)(ws + alloc(BTC * 2));
    _Float16* VC  = (_Float16*)(ws + alloc(BTC * 2));
    _Float16* VT  = (_Float16*)(ws + alloc((size_t)Bn * Cn * Tn * 2));
    float*    LOG = (float*)   (ws + alloc((size_t)Bn * Tn * Tn * 4));
    float*    AO  = (float*)   (ws + alloc(BTC * 4));
    float*    O0  = (float*)   (ws + alloc(BTC * 4));
    float*    O1  = (float*)   (ws + alloc(BTC * 4));
    float*    O2  = (float*)   (ws + alloc(BTC * 4));

    auto f2h = [&](const float* s, _Float16* d, size_t n) {
        f2h_kernel<<<dim3((unsigned)((n + 255) / 256)), 256, 0, stream>>>(s, d, n);
    };
    auto gemm = [&](const _Float16* A, int sA, long aB, const _Float16* Bm, int sB, long bB,
                    float* Yf, _Float16* Yh, int sY, long yB, const float* bias,
                    float cs, const float* tp, int M, int N, int K, int batches) {
        int tiles = (M / 16) * (N / 64);
        gemm_wmma_f16<<<dim3((tiles + 7) / 8, 1, batches), 256, 0, stream>>>(
            A, sA, aB, Bm, sB, bB, Yf, Yh, sY, yB, bias, cs, tp, M, N, K);
    };

    // --- precision conversions ---
    f2h(x, XH, BTC);
    for (int i = 0; i < 9; ++i) f2h(W[i], WH + (size_t)i * CC, CC);

    // --- downsample conv -> XDH (async-to-LDS staging) ---
    conv_down_kernel<<<dim3(Cn / 256, Tdn, Bn), 256, 0, stream>>>(x, dw, db, XDH);

    // --- projections ---
    const int MT = Bn * Tn, MTd = Bn * Tdn;
    gemm(XH,  Cn, 0, WH + 0 * CC, Cn, 0, nullptr, QCR, Cn, 0, Bi[0], 1.f, nullptr, MT,  Cn, Cn, 1);
    gemm(XDH, Cn, 0, WH + 1 * CC, Cn, 0, nullptr, KCR, Cn, 0, Bi[1], 1.f, nullptr, MTd, Cn, Cn, 1);
    gemm(XDH, Cn, 0, WH + 2 * CC, Cn, 0, nullptr, VCR, Cn, 0, Bi[2], 1.f, nullptr, MTd, Cn, Cn, 1);
    gemm(XH,  Cn, 0, WH + 3 * CC, Cn, 0, nullptr, QD,  Cn, 0, Bi[3], 1.f, nullptr, MT,  Cn, Cn, 1);
    gemm(XH,  Cn, 0, WH + 4 * CC, Cn, 0, nullptr, KD,  Cn, 0, Bi[4], 1.f, nullptr, MT,  Cn, Cn, 1);
    gemm(XH,  Cn, 0, WH + 5 * CC, Cn, 0, nullptr, VD,  Cn, 0, Bi[5], 1.f, nullptr, MT,  Cn, Cn, 1);
    gemm(XH,  Cn, 0, WH + 6 * CC, Cn, 0, nullptr, QC,  Cn, 0, Bi[6], 1.f, nullptr, MT,  Cn, Cn, 1);
    gemm(XH,  Cn, 0, WH + 7 * CC, Cn, 0, nullptr, KC,  Cn, 0, Bi[7], 1.f, nullptr, MT,  Cn, Cn, 1);
    gemm(XH,  Cn, 0, WH + 8 * CC, Cn, 0, nullptr, VC,  Cn, 0, Bi[8], 1.f, nullptr, MT,  Cn, Cn, 1);

    // --- cosine branch normalization (in place on QC/KC) ---
    l2norm_rows_kernel<<<(MT + 7) / 8, 256, 0, stream>>>(QC, MT);
    l2norm_rows_kernel<<<(MT + 7) / 8, 256, 0, stream>>>(KC, MT);

    const float invSqrtC = 0.044194173824159216f; // 1/sqrt(512)

    struct Branch { _Float16 *q, *k, *v; int S; const float* tp; float cs; float* out; int li; };
    Branch br[3] = {
        {QCR, KCR, VCR, Tdn, nullptr, invSqrtC, O0, 0},
        {QD,  KD,  VD,  Tn,  nullptr, invSqrtC, O1, 1},
        {QC,  KC,  VC,  Tn,  temp,    1.0f,     O2, 2},
    };

    for (int i = 0; i < 3; ++i) {
        int S = br[i].S;
        // logits = scale * Q @ K^T  -> f32 [B, T, S]
        gemm(br[i].q, Cn, (long)Tn * Cn, br[i].k, Cn, (long)S * Cn,
             LOG, nullptr, S, (long)Tn * S, nullptr, br[i].cs, br[i].tp, Tn, S, Cn, Bn);
        // softmax rows, f16 probs written in place
        softmax_rows_kernel<<<Bn * Tn, 256, 0, stream>>>(LOG, S);
        // transpose V -> VT [B, C, S]
        transpose_f16_kernel<<<dim3(S / 32, Cn / 32, Bn), dim3(32, 8), 0, stream>>>(br[i].v, VT, S);
        // O = P @ V  (A = f16 probs with row stride 2S over the f32 buffer)
        gemm((const _Float16*)LOG, 2 * S, (long)Tn * 2 * S, VT, S, (long)Cn * S,
             AO, nullptr, Cn, (long)Tn * Cn, nullptr, 1.0f, nullptr, Tn, Cn, S, Bn);
        // LayerNorm
        ln_rows_kernel<<<(MT + 7) / 8, 256, 0, stream>>>(AO, br[i].out, ln_g[br[i].li], ln_b[br[i].li], MT);
    }

    // --- gating fusion -> d_out ---
    gating_kernel<<<Bn * Tn, 128, 0, stream>>>(O0, O1, O2, gw, gb, zout);
}
